// MultiHeadSelfAttention_26414048870481
// MI455X (gfx1250) — matrix operand order repaired and checked by reading, not compiled, not run on hardware
//
#include <hip/hip_runtime.h>
#include <hip/hip_bf16.h>

// MHA: B=2, S=2048, D=1024, H=16, Dk=64.  All heavy math via
// v_wmma_f32_16x16x32_bf16 (CDNA5 wave32 WMMA).  Flash-style attention so the
// 512MB score matrix never hits HBM.

#define S_LEN  2048
#define DMODEL 1024
#define NHEADS 16
#define DKH    64
#define NBATCH 2
#define MTOT   (NBATCH * S_LEN)   // 4096 rows

typedef __bf16 bf16_t;
typedef __attribute__((ext_vector_type(16))) __bf16 v16bf;
typedef __attribute__((ext_vector_type(8)))  float  v8f;
typedef unsigned int u32x4 __attribute__((ext_vector_type(4)));

union FragU { v16bf v; u32x4 q[2]; };

// A-operand fragment (16x32 bf16): lane (l&15)=row m; elements 0..7 hold
// K = khalf*8 + j, elements 8..15 hold K = 16 + khalf*8 + j  (khalf = l>>4).
__device__ __forceinline__ v16bf load_a_frag(const bf16_t* __restrict__ row,
                                             int k0, int khalf) {
  FragU f;
  f.q[0] = *reinterpret_cast<const u32x4*>(row + k0 + khalf * 8);
  f.q[1] = *reinterpret_cast<const u32x4*>(row + k0 + 16 + khalf * 8);
  return f.v;
}

// B-operand fragment (32x16 bf16): lane (l&15)=col n; elements 0..7 hold
// K = khalf*8 + j, elements 8..15 hold K = 16 + khalf*8 + j (two 128-bit loads).
__device__ __forceinline__ v16bf load_b_frag(const bf16_t* __restrict__ row,
                                             int k0, int khalf) {
  FragU f;
  f.q[0] = *reinterpret_cast<const u32x4*>(row + k0 + khalf * 8);
  f.q[1] = *reinterpret_cast<const u32x4*>(row + k0 + 16 + khalf * 8);
  return f.v;
}

__device__ __forceinline__ v8f wmma_bf16(v16bf a, v16bf b, v8f c) {
  return __builtin_amdgcn_wmma_f32_16x16x32_bf16(false, a, false, b,
                                                 (short)0, c, false, false);
}

__global__ void f32_to_bf16_kernel(const float* __restrict__ src,
                                   bf16_t* __restrict__ dst, int n) {
  int i = blockIdx.x * blockDim.x + threadIdx.x;
  if (i < n) dst[i] = (bf16_t)src[i];
}

// NT GEMM: Y[m,n] = sum_k A[m,k] * W[n,k] + bias[n].  M=4096, N=1024, K=1024.
// Block = 256 threads = 8 waves; wave computes a 16x64 tile (4 accumulators).
// MODE 0: Q  -> bf16 [b,h,s,64], scaled by 1/sqrt(Dk)=0.125
// MODE 1: K  -> bf16 [b,h,s,64]
// MODE 2: V  -> bf16 [b,h,64,s]   (transposed for contiguous P@V B-fragments)
// MODE 3: Out-> f32  [m,n] (d_out)
template <int MODE>
__global__ __launch_bounds__(256)
void gemm_nt_wmma(const bf16_t* __restrict__ A, const bf16_t* __restrict__ W,
                  const float* __restrict__ bias, void* __restrict__ out) {
  const int lane  = threadIdx.x & 31;
  const int wave  = threadIdx.x >> 5;
  const int lm    = lane & 15;
  const int khalf = lane >> 4;
  const int n0 = blockIdx.x * 64;
  const int m0 = blockIdx.y * 128 + wave * 16;

  const bf16_t* arow = A + (size_t)(m0 + lm) * DMODEL;
  v8f acc[4] = {};

  for (int k0 = 0; k0 < DMODEL; k0 += 32) {
    v16bf af = load_a_frag(arow, k0, khalf);
#pragma unroll
    for (int t = 0; t < 4; ++t) {
      v16bf bfr = load_b_frag(W + (size_t)(n0 + t * 16 + lm) * DMODEL, k0, khalf);
      acc[t] = wmma_bf16(af, bfr, acc[t]);
    }
  }

  // C/D layout: VGPR v <-> row (v + 8*khalf), col = lane&15.
#pragma unroll
  for (int t = 0; t < 4; ++t) {
#pragma unroll
    for (int v = 0; v < 8; ++v) {
      const int m = m0 + v + 8 * khalf;
      const int n = n0 + t * 16 + lm;
      float val = acc[t][v] + bias[n];
      if (MODE == 0) val *= 0.125f;   // 1/sqrt(64), exact power of two
      if (MODE == 3) {
        reinterpret_cast<float*>(out)[(size_t)m * DMODEL + n] = val;
      } else {
        const int b = m >> 11, s = m & (S_LEN - 1);
        const int h = n >> 6,  d = n & (DKH - 1);
        size_t idx;
        if (MODE == 2)
          idx = (((size_t)(b * NHEADS + h)) * DKH + d) * S_LEN + s;
        else
          idx = (((size_t)(b * NHEADS + h)) * S_LEN + s) * DKH + d;
        reinterpret_cast<bf16_t*>(out)[idx] = (bf16_t)val;
      }
    }
  }
}

// Flash attention: grid (B*H, S/128); 8 waves, each owns 16 Q rows.
// Per 32-key block: 4 WMMAs for S = Q*K^T, online softmax (shfl_xor butterfly
// inside 16-lane halves), P restaged via per-wave LDS into A-layout,
// 4 WMMAs for O += P*V.  Writes O as bf16 [b,s,h*64+d].
__global__ __launch_bounds__(256)
void flash_attn_wmma(const bf16_t* __restrict__ Q, const bf16_t* __restrict__ K,
                     const bf16_t* __restrict__ Vt, bf16_t* __restrict__ O) {
  const int lane  = threadIdx.x & 31;
  const int wave  = threadIdx.x >> 5;
  const int lm    = lane & 15;
  const int khalf = lane >> 4;
  const int bh = blockIdx.x;
  const int b = bh >> 4, h = bh & 15;
  const int q0 = blockIdx.y * 128 + wave * 16;

  const bf16_t* Qh = Q  + (size_t)bh * S_LEN * DKH;
  const bf16_t* Kh = K  + (size_t)bh * S_LEN * DKH;
  const bf16_t* Vh = Vt + (size_t)bh * DKH * S_LEN;

  __shared__ bf16_t pbuf[8][16][34];   // per-wave 16x32 P tile, padded rows

  const bf16_t* qrow = Qh + (size_t)(q0 + lm) * DKH;
  v16bf qf0 = load_a_frag(qrow, 0, khalf);
  v16bf qf1 = load_a_frag(qrow, 32, khalf);

  v8f oacc[4] = {};
  float mrow[8], lrow[8];
#pragma unroll
  for (int v = 0; v < 8; ++v) { mrow[v] = -3.0e38f; lrow[v] = 0.0f; }

  for (int kb = 0; kb < S_LEN; kb += 32) {
    if (kb + 32 < S_LEN) {  // lookahead -> global_prefetch_b8
      __builtin_prefetch(Kh + (size_t)(kb + 32 + lane) * DKH, 0, 0);
      __builtin_prefetch(Vh + (size_t)(lane & (DKH - 1)) * S_LEN + kb + 32, 0, 0);
    }
    // scores for keys kb..kb+15 (s0) and kb+16..kb+31 (s1); Dk=64 -> 2 k-steps
    v8f s0 = {}, s1 = {};
    {
      v16bf kf;
      kf = load_b_frag(Kh + (size_t)(kb + lm) * DKH, 0, khalf);
      s0 = wmma_bf16(qf0, kf, s0);
      kf = load_b_frag(Kh + (size_t)(kb + lm) * DKH, 32, khalf);
      s0 = wmma_bf16(qf1, kf, s0);
      kf = load_b_frag(Kh + (size_t)(kb + 16 + lm) * DKH, 0, khalf);
      s1 = wmma_bf16(qf0, kf, s1);
      kf = load_b_frag(Kh + (size_t)(kb + 16 + lm) * DKH, 32, khalf);
      s1 = wmma_bf16(qf1, kf, s1);
    }
    // online softmax: row (v + 8*khalf) lives entirely in one 16-lane half
#pragma unroll
    for (int v = 0; v < 8; ++v) {
      float tmax = fmaxf(s0[v], s1[v]);
      tmax = fmaxf(tmax, __shfl_xor(tmax, 1, 32));
      tmax = fmaxf(tmax, __shfl_xor(tmax, 2, 32));
      tmax = fmaxf(tmax, __shfl_xor(tmax, 4, 32));
      tmax = fmaxf(tmax, __shfl_xor(tmax, 8, 32));
      const float mnew = fmaxf(mrow[v], tmax);
      const float sc = __expf(mrow[v] - mnew);
      mrow[v] = mnew;
      const float p0 = __expf(s0[v] - mnew);
      const float p1 = __expf(s1[v] - mnew);
      float ts = p0 + p1;
      ts += __shfl_xor(ts, 1, 32);
      ts += __shfl_xor(ts, 2, 32);
      ts += __shfl_xor(ts, 4, 32);
      ts += __shfl_xor(ts, 8, 32);
      lrow[v] = lrow[v] * sc + ts;
#pragma unroll
      for (int t = 0; t < 4; ++t) oacc[t][v] *= sc;
      // C-layout -> LDS row-major P (bf16)
      pbuf[wave][v + 8 * khalf][lm]      = (bf16_t)p0;
      pbuf[wave][v + 8 * khalf][16 + lm] = (bf16_t)p1;
    }
    // Reload P in A-layout (same-wave LDS ops are in-order; no barrier needed)
    v16bf pa = {};
#pragma unroll
    for (int j = 0; j < 8; ++j) {
      pa[j]     = pbuf[wave][lm][khalf * 8 + j];
      pa[8 + j] = pbuf[wave][lm][16 + khalf * 8 + j];
    }
    // O += P (16x32) * V (32 x 64); Vt gives contiguous B-fragments
#pragma unroll
    for (int t = 0; t < 4; ++t) {
      v16bf vf = load_b_frag(Vh + (size_t)(t * 16 + lm) * S_LEN, kb, khalf);
      oacc[t] = wmma_bf16(pa, vf, oacc[t]);
    }
  }

  // epilogue: O[b, q, h*64 + d] = acc / l
#pragma unroll
  for (int t = 0; t < 4; ++t) {
#pragma unroll
    for (int v = 0; v < 8; ++v) {
      const int qr = q0 + v + 8 * khalf;
      const int d  = t * 16 + lm;
      const float val = oacc[t][v] / lrow[v];
      O[((size_t)(b * S_LEN + qr)) * DMODEL + h * DKH + d] = (bf16_t)val;
    }
  }
}

extern "C" void kernel_launch(void* const* d_in, const int* in_sizes, int n_in,
                              void* d_out, int out_size, void* d_ws, size_t ws_size,
                              hipStream_t stream) {
  (void)in_sizes; (void)n_in; (void)out_size; (void)ws_size;
  const float* x  = (const float*)d_in[0];
  const float* Wq = (const float*)d_in[1];
  const float* bq = (const float*)d_in[2];
  const float* Wk = (const float*)d_in[3];
  const float* bk = (const float*)d_in[4];
  const float* Wv = (const float*)d_in[5];
  const float* bv = (const float*)d_in[6];
  const float* Wo = (const float*)d_in[7];
  const float* bo = (const float*)d_in[8];

  // Workspace layout (bf16 elements), 48 MiB total:
  bf16_t* xb  = (bf16_t*)d_ws;                 // 4M  : x as bf16 [4096,1024]
  bf16_t* wqb = xb  + (size_t)MTOT * DMODEL;   // 1M
  bf16_t* wkb = wqb + (size_t)DMODEL * DMODEL; // 1M
  bf16_t* wvb = wkb + (size_t)DMODEL * DMODEL; // 1M
  bf16_t* wob = wvb + (size_t)DMODEL * DMODEL; // 1M
  bf16_t* Qb  = wob + (size_t)DMODEL * DMODEL; // 4M : [b,h,s,64], pre-scaled
  bf16_t* Kb  = Qb  + (size_t)MTOT * DMODEL;   // 4M : [b,h,s,64]
  bf16_t* Vtb = Kb  + (size_t)MTOT * DMODEL;   // 4M : [b,h,64,s]
  bf16_t* Ob  = Vtb + (size_t)MTOT * DMODEL;   // 4M : [b,s,1024]

  const int nx = MTOT * DMODEL;       // 4,194,304
  const int nw = DMODEL * DMODEL;     // 1,048,576
  f32_to_bf16_kernel<<<(nx + 255) / 256, 256, 0, stream>>>(x,  xb,  nx);
  f32_to_bf16_kernel<<<(nw + 255) / 256, 256, 0, stream>>>(Wq, wqb, nw);
  f32_to_bf16_kernel<<<(nw + 255) / 256, 256, 0, stream>>>(Wk, wkb, nw);
  f32_to_bf16_kernel<<<(nw + 255) / 256, 256, 0, stream>>>(Wv, wvb, nw);
  f32_to_bf16_kernel<<<(nw + 255) / 256, 256, 0, stream>>>(Wo, wob, nw);

  dim3 ggrid(DMODEL / 64, MTOT / 128);   // (16, 32)
  gemm_nt_wmma<0><<<ggrid, 256, 0, stream>>>(xb, wqb, bq, (void*)Qb);
  gemm_nt_wmma<1><<<ggrid, 256, 0, stream>>>(xb, wkb, bk, (void*)Kb);
  gemm_nt_wmma<2><<<ggrid, 256, 0, stream>>>(xb, wvb, bv, (void*)Vtb);

  dim3 fgrid(NBATCH * NHEADS, S_LEN / 128);  // (32, 16)
  flash_attn_wmma<<<fgrid, 256, 0, stream>>>(Qb, Kb, Vtb, Ob);

  gemm_nt_wmma<3><<<ggrid, 256, 0, stream>>>(Ob, wob, bo, d_out);
}
